// Attention_7679401525299
// MI455X (gfx1250) — matrix operand order repaired and checked by reading, not compiled, not run on hardware
//
#include <hip/hip_runtime.h>

// ---------------------------------------------------------------------------
// Types for CDNA5 WMMA (wave32): v_wmma_f32_16x16x32_bf16
// ---------------------------------------------------------------------------
typedef __attribute__((ext_vector_type(16))) __bf16 v16bf;
typedef __attribute__((ext_vector_type(8)))  __bf16 v8bf;
typedef __attribute__((ext_vector_type(8)))  float  v8f;
typedef __attribute__((ext_vector_type(4)))  int    v4i;

#define DEV static __device__ __forceinline__

// gfx1250 async global->LDS path (ASYNCcnt), guarded so either toolchain compiles.
#if defined(__AMDGCN__) && __has_builtin(__builtin_amdgcn_global_load_async_to_lds_b128)
#define HAVE_ASYNC_LDS 1
#else
#define HAVE_ASYNC_LDS 0
#endif
#if defined(__AMDGCN__) && __has_builtin(__builtin_amdgcn_s_wait_asynccnt)
#define HAVE_ASYNCCNT 1
#else
#define HAVE_ASYNCCNT 0
#endif

// Copy 16 bytes (8 bf16) global -> LDS. Async (no VGPR bounce) when available.
DEV void stage16(unsigned short* l, const unsigned short* g) {
#if HAVE_ASYNC_LDS
    __builtin_amdgcn_global_load_async_to_lds_b128(
        (__attribute__((address_space(1))) v4i*)g,
        (__attribute__((address_space(3))) v4i*)l,
        0, 0);
#else
    *(uint4*)l = *(const uint4*)g;
#endif
}
DEV void stage_wait() {
#if HAVE_ASYNC_LDS && HAVE_ASYNCCNT
    __builtin_amdgcn_s_wait_asynccnt(0);
#endif
}

DEV unsigned short f2bf(float f) {
    union { float f; unsigned u; } v; v.f = f;
    unsigned r = (v.u + 0x7FFFu + ((v.u >> 16) & 1u)) >> 16;  // RNE
    return (unsigned short)r;
}
DEV unsigned pack2(float a, float b) {
    return (unsigned)f2bf(a) | ((unsigned)f2bf(b) << 16);
}
DEV v8f vzero8() { v8f z = {0.f,0.f,0.f,0.f,0.f,0.f,0.f,0.f}; return z; }

// A-matrix fragment (16x32 bf16, MxK), ISA 7.12.2 layout:
// lane L(0..15): row M=L, K {0..7,16..23}; lane L+16: row M=L, K {8..15,24..31}
DEV v16bf load_frag_a(const unsigned short* base, int ld) {
    int lane = threadIdx.x & 31;
    int row  = lane & 15;
    int hi8  = (lane >> 4) << 3;  // 0 or 8
    v8bf lo = *(const v8bf*)(base + row * ld + hi8);
    v8bf hh = *(const v8bf*)(base + row * ld + 16 + hi8);
    return __builtin_shufflevector(lo, hh, 0,1,2,3,4,5,6,7,8,9,10,11,12,13,14,15);
}

// B-matrix fragment (32x16 bf16, KxN), stored in LDS as [N rows][K cols]:
// lanes 0..15: N=lane, K {0..7,16..23} ; lanes 16..31: N=lane-16, K {8..15,24..31}
DEV v16bf load_frag_b(const unsigned short* base, int ld) {
    int lane = threadIdx.x & 31;
    int n    = lane & 15;
    int hi8  = (lane >> 4) << 3;  // 0 or 8
    v8bf lo = *(const v8bf*)(base + n * ld + hi8);
    v8bf hh = *(const v8bf*)(base + n * ld + 16 + hi8);
    return __builtin_shufflevector(lo, hh, 0,1,2,3,4,5,6,7,8,9,10,11,12,13,14,15);
}

DEV v8f wmma_bf16(v16bf a, v16bf b, v8f c) {
    return __builtin_amdgcn_wmma_f32_16x16x32_bf16(false, a, false, b, (short)0, c,
                                                   false, false);
}

// ---------------------------------------------------------------------------
// Constants
// ---------------------------------------------------------------------------
static constexpr int BATCH  = 4;
static constexpr int SEQ    = 2048;
static constexpr int DMODEL = 1024;
static constexpr int NHEADS = 16;
static constexpr int DHEAD  = 64;

// ---------------------------------------------------------------------------
// GEMM: C[M,N] = A[M,K] @ W[N,K]^T   (nn.Linear convention, both row-major-K)
// A is f32 (A_BF16=0) or bf16 (A_BF16=1); W always f32, converted to bf16.
// LAYOUT: 0 = f32 plain [m][n]
//         1 = bf16 head-major      [(b*H+h)*S + s][dh]  (Q, K for attention)
//         2 = bf16 head-transposed [(b*H+h)*Dh + dh][s] (V^T for PV WMMA)
// Tiling: 128x128x64, 256 threads = 8 waves in 4x2 grid, 32x64 per wave.
// 16 WMMAs per barrier pair.
// ---------------------------------------------------------------------------
template<int A_BF16, int LAYOUT>
__global__ __launch_bounds__(256)
void gemm_wmma(const void* __restrict__ Ap, const float* __restrict__ W,
               void* __restrict__ Cp, int M, int N, int K)
{
    __shared__ __align__(32) unsigned short As[128 * 64];   // 16 KB
    __shared__ __align__(32) unsigned short Bs[128 * 64];   // 16 KB

    const int tid  = threadIdx.x;
    const int wave = tid >> 5;
    const int wm   = wave & 3;   // 0..3 : 32-row band
    const int wn   = wave >> 2;  // 0..1 : 64-col band
    const int m0   = blockIdx.x * 128;
    const int n0   = blockIdx.y * 128;

    const int lrow = tid >> 1;          // 0..127
    const int lcol = (tid & 1) << 5;    // 0 or 32 (elements)

    v8f acc[2][4];
    for (int i = 0; i < 2; ++i)
        for (int j = 0; j < 4; ++j) acc[i][j] = vzero8();

    const float*          Af = (const float*)Ap;
    const unsigned short* Ab = (const unsigned short*)Ap;

    for (int k0 = 0; k0 < K; k0 += 64) {
        // ---- stage A tile (128x64) into LDS as bf16 ----
        if (A_BF16) {
            const unsigned short* g = Ab + (size_t)(m0 + lrow) * K + k0 + lcol;
            unsigned short*       s = As + lrow * 64 + lcol;
            stage16(s, g); stage16(s + 8, g + 8);
            stage16(s + 16, g + 16); stage16(s + 24, g + 24);
        } else {
            const float4* g = (const float4*)(Af + (size_t)(m0 + lrow) * K + k0 + lcol);
            uint4* s = (uint4*)(As + lrow * 64 + lcol);
            #pragma unroll
            for (int u = 0; u < 4; ++u) {
                float4 a = g[2 * u], b = g[2 * u + 1];
                s[u] = make_uint4(pack2(a.x, a.y), pack2(a.z, a.w),
                                  pack2(b.x, b.y), pack2(b.z, b.w));
            }
        }
        // ---- stage B tile (128x64 of W, f32 -> bf16) ----
        {
            const float4* g = (const float4*)(W + (size_t)(n0 + lrow) * K + k0 + lcol);
            uint4* s = (uint4*)(Bs + lrow * 64 + lcol);
            #pragma unroll
            for (int u = 0; u < 4; ++u) {
                float4 a = g[2 * u], b = g[2 * u + 1];
                s[u] = make_uint4(pack2(a.x, a.y), pack2(a.z, a.w),
                                  pack2(b.x, b.y), pack2(b.z, b.w));
            }
        }
        // prefetch next k-tile (global_prefetch_b8)
        if (k0 + 64 < K) {
            const char* pa = (const char*)Ap +
                ((size_t)(m0 + lrow) * K + k0 + 64 + lcol) * (A_BF16 ? 2 : 4);
            __builtin_prefetch(pa, 0, 1);
            __builtin_prefetch(W + (size_t)(n0 + lrow) * K + k0 + 64 + lcol, 0, 1);
        }
        if (A_BF16) stage_wait();
        __syncthreads();

        v16bf af[2][2];
        af[0][0] = load_frag_a(As + (wm * 32 +  0) * 64,      64);
        af[0][1] = load_frag_a(As + (wm * 32 +  0) * 64 + 32, 64);
        af[1][0] = load_frag_a(As + (wm * 32 + 16) * 64,      64);
        af[1][1] = load_frag_a(As + (wm * 32 + 16) * 64 + 32, 64);
        #pragma unroll
        for (int j = 0; j < 4; ++j) {
            v16bf b0 = load_frag_b(Bs + (wn * 64 + j * 16) * 64,      64);
            v16bf b1 = load_frag_b(Bs + (wn * 64 + j * 16) * 64 + 32, 64);
            acc[0][j] = wmma_bf16(af[0][0], b0, acc[0][j]);
            acc[0][j] = wmma_bf16(af[0][1], b1, acc[0][j]);
            acc[1][j] = wmma_bf16(af[1][0], b0, acc[1][j]);
            acc[1][j] = wmma_bf16(af[1][1], b1, acc[1][j]);
        }
        __syncthreads();
    }

    // ---- store: C/D layout = row r + 8*(lane>>4), col lane&15 ----
    const int lane = tid & 31;
    const int ln   = lane & 15;
    const int hi8  = (lane >> 4) * 8;
    for (int i = 0; i < 2; ++i)
        for (int j = 0; j < 4; ++j)
            #pragma unroll
            for (int r = 0; r < 8; ++r) {
                int m = m0 + wm * 32 + i * 16 + r + hi8;
                int n = n0 + wn * 64 + j * 16 + ln;
                float val = acc[i][j][r];
                if (LAYOUT == 0) {
                    ((float*)Cp)[(size_t)m * N + n] = val;
                } else if (LAYOUT == 1) {
                    int b = m >> 11, s = m & (SEQ - 1);
                    int h = n >> 6,  d = n & (DHEAD - 1);
                    ((unsigned short*)Cp)
                        [(((size_t)(b * NHEADS + h) * SEQ) + s) * DHEAD + d] = f2bf(val);
                } else {
                    int b = m >> 11, s = m & (SEQ - 1);
                    int h = n >> 6,  d = n & (DHEAD - 1);
                    ((unsigned short*)Cp)
                        [(((size_t)(b * NHEADS + h) * DHEAD) + d) * SEQ + s] = f2bf(val);
                }
            }
}

// ---------------------------------------------------------------------------
// Flash attention: one WG = 128 query rows of one (b,h); 8 waves x 16 q-rows.
// Q: [bh][s][dh] bf16, K: [bh][s][dh] bf16, Vt: [bh][dh][s] bf16.
// Online softmax, scale = 1/sqrt(DMODEL) = 1/32 (folded into log2 domain).
// Out: bf16 [b][s][h*64+d]  (ready to be the A of the output projection).
// ---------------------------------------------------------------------------
__global__ __launch_bounds__(256)
void flash_attn(const unsigned short* __restrict__ Q,
                const unsigned short* __restrict__ Kh,
                const unsigned short* __restrict__ Vt,
                unsigned short* __restrict__ Out)
{
    __shared__ __align__(32) unsigned short Qs[128 * 64];    // 16 KB
    __shared__ __align__(32) unsigned short Ks[64 * 64];     //  8 KB
    __shared__ __align__(32) unsigned short Vs[64 * 64];     //  8 KB (V^T tile)
    __shared__ __align__(32) unsigned short Ps[8 * 16 * 64]; // 16 KB

    const int tid  = threadIdx.x;
    const int wave = tid >> 5;
    const int lane = tid & 31;
    const int ln   = lane & 15;
    const int hi   = lane >> 4;
    const int bh   = blockIdx.y;
    const int q0   = blockIdx.x * 128;

    const unsigned short* Qg = Q  + ((size_t)bh * SEQ + q0) * DHEAD;
    const unsigned short* Kg = Kh + (size_t)bh * SEQ * DHEAD;
    const unsigned short* Vg = Vt + (size_t)bh * DHEAD * SEQ;

    {   // load Q tile 128x64 bf16 (async -> LDS when available)
        int row = tid >> 1, cb = (tid & 1) * 32;
        const unsigned short* g = Qg + (size_t)row * DHEAD + cb;
        unsigned short*       s = Qs + row * 64 + cb;
        stage16(s, g); stage16(s + 8, g + 8);
        stage16(s + 16, g + 16); stage16(s + 24, g + 24);
    }
    stage_wait();
    __syncthreads();

    v16bf qf0 = load_frag_a(Qs + wave * 16 * 64,      64);  // dh 0..31
    v16bf qf1 = load_frag_a(Qs + wave * 16 * 64 + 32, 64);  // dh 32..63

    float mstate[8], lstate[8];
    v8f oacc[4];
    #pragma unroll
    for (int r = 0; r < 8; ++r) { mstate[r] = -1e30f; lstate[r] = 0.f; }
    for (int o = 0; o < 4; ++o) oacc[o] = vzero8();

    const float scale2 = 0.03125f * 1.44269504f;   // (1/sqrt(1024)) * log2(e)
    unsigned short* Pw = Ps + wave * 16 * 64;

    for (int kt = 0; kt < SEQ; kt += 64) {
        {   // K tile [64 keys][64 dh], V^T tile [64 dh][64 keys]
            int row = tid >> 2, cb = (tid & 3) * 16;
            const unsigned short* gk = Kg + (size_t)(kt + row) * DHEAD + cb;
            unsigned short*       sk = Ks + row * 64 + cb;
            stage16(sk, gk); stage16(sk + 8, gk + 8);
            const unsigned short* gv = Vg + (size_t)row * SEQ + kt + cb;
            unsigned short*       sv = Vs + row * 64 + cb;
            stage16(sv, gv); stage16(sv + 8, gv + 8);
        }
        stage_wait();
        __syncthreads();

        // ---- scores: S(16x64) = Q(16x64dh) @ K_tile^T ----
        v8f sacc[4];
        #pragma unroll
        for (int n = 0; n < 4; ++n) {
            sacc[n] = vzero8();
            v16bf b0 = load_frag_b(Ks + n * 16 * 64,      64);  // dh 0..31
            v16bf b1 = load_frag_b(Ks + n * 16 * 64 + 32, 64);  // dh 32..63
            sacc[n] = wmma_bf16(qf0, b0, sacc[n]);
            sacc[n] = wmma_bf16(qf1, b1, sacc[n]);
        }

        // ---- online softmax (log2 domain) ----
        #pragma unroll
        for (int r = 0; r < 8; ++r) {
            float s0 = sacc[0][r] * scale2, s1 = sacc[1][r] * scale2;
            float s2 = sacc[2][r] * scale2, s3 = sacc[3][r] * scale2;
            float v = fmaxf(fmaxf(s0, s1), fmaxf(s2, s3));
            #pragma unroll
            for (int off = 1; off < 16; off <<= 1)
                v = fmaxf(v, __shfl_xor(v, off, 32));
            float mnew = fmaxf(mstate[r], v);
            float corr = exp2f(mstate[r] - mnew);
            mstate[r] = mnew;
            float p0 = exp2f(s0 - mnew), p1 = exp2f(s1 - mnew);
            float p2 = exp2f(s2 - mnew), p3 = exp2f(s3 - mnew);
            sacc[0][r] = p0; sacc[1][r] = p1; sacc[2][r] = p2; sacc[3][r] = p3;
            float ps = p0 + p1 + p2 + p3;
            #pragma unroll
            for (int off = 1; off < 16; off <<= 1)
                ps += __shfl_xor(ps, off, 32);
            lstate[r] = lstate[r] * corr + ps;
            for (int o = 0; o < 4; ++o) oacc[o][r] *= corr;
        }

        // ---- transpose P into A-fragment layout via per-wave LDS slab ----
        #pragma unroll
        for (int n = 0; n < 4; ++n)
            #pragma unroll
            for (int r = 0; r < 8; ++r)
                Pw[(r + 8 * hi) * 64 + n * 16 + ln] = f2bf(sacc[n][r]);
        __syncthreads();

        // ---- O += P(16x64k) @ V(64k x 64dh)  (Vt rows are dh, cols k) ----
        v16bf pf0 = load_frag_a(Pw,      64);   // keys 0..31
        v16bf pf1 = load_frag_a(Pw + 32, 64);   // keys 32..63
        #pragma unroll
        for (int o = 0; o < 4; ++o) {
            v16bf b0 = load_frag_b(Vs + o * 16 * 64,      64);
            v16bf b1 = load_frag_b(Vs + o * 16 * 64 + 32, 64);
            oacc[o] = wmma_bf16(pf0, b0, oacc[o]);
            oacc[o] = wmma_bf16(pf1, b1, oacc[o]);
        }
        __syncthreads();
    }

    // ---- finalize: O /= l, store bf16 to [b][s][h*64+d] ----
    const int b = bh >> 4, h = bh & 15;
    for (int o = 0; o < 4; ++o)
        #pragma unroll
        for (int r = 0; r < 8; ++r) {
            int qrow = q0 + wave * 16 + r + 8 * hi;
            int col  = h * DHEAD + o * 16 + ln;
            float val = oacc[o][r] / lstate[r];
            Out[((size_t)(b * SEQ + qrow)) * DMODEL + col] = f2bf(val);
        }
}

// ---------------------------------------------------------------------------
// Launcher
// ---------------------------------------------------------------------------
extern "C" void kernel_launch(void* const* d_in, const int* in_sizes, int n_in,
                              void* d_out, int out_size, void* d_ws, size_t ws_size,
                              hipStream_t stream)
{
    (void)in_sizes; (void)n_in; (void)out_size; (void)ws_size;

    const float* q  = (const float*)d_in[0];
    const float* k  = (const float*)d_in[1];
    const float* v  = (const float*)d_in[2];
    const float* wq = (const float*)d_in[3];
    const float* wk = (const float*)d_in[4];
    const float* wv = (const float*)d_in[5];
    const float* wo = (const float*)d_in[6];

    const int M = BATCH * SEQ;          // 8192
    const int D = DMODEL;               // 1024
    const size_t planeElems = (size_t)M * D;  // bf16 elements per plane

    unsigned short* qb = (unsigned short*)d_ws;
    unsigned short* kb = qb + planeElems;
    unsigned short* vb = kb + planeElems;
    unsigned short* ab = vb + planeElems;

    dim3 gg(M / 128, D / 128);   // 64 x 8
    dim3 bb(256);

    // Q/K/V projections (f32 in, bf16 head-layout out)
    gemm_wmma<0, 1><<<gg, bb, 0, stream>>>(q, wq, qb, M, D, D);
    gemm_wmma<0, 1><<<gg, bb, 0, stream>>>(k, wk, kb, M, D, D);
    gemm_wmma<0, 2><<<gg, bb, 0, stream>>>(v, wv, vb, M, D, D);

    // flash attention: grid (S/128, B*H)
    flash_attn<<<dim3(SEQ / 128, BATCH * NHEADS), bb, 0, stream>>>(qb, kb, vb, ab);

    // output projection (bf16 in, f32 plain out)
    gemm_wmma<1, 0><<<gg, bb, 0, stream>>>(ab, wo, (float*)d_out, M, D, D);
}
